// MambaBlock_21577915695280
// MI455X (gfx1250) — matrix operand order repaired and checked by reading, not compiled, not run on hardware
//
#include <hip/hip_runtime.h>
#include <hip/hip_bf16.h>

// ---- problem constants (Mamba block) ----
#define ML    4096   // B*L rows
#define BB    2
#define LLEN  2048
#define DM    1024   // d_model
#define DI    2048   // d_inner
#define PROJW 4096   // 2*d_inner
#define NS    16     // d_state
#define DTR   64     // dt_rank
#define PDIM  96     // dt_rank + 2*N

#define AS1 __attribute__((address_space(1)))
#define AS3 __attribute__((address_space(3)))

typedef __attribute__((ext_vector_type(16))) __bf16 bfx16;
typedef __attribute__((ext_vector_type(8)))  __bf16 bfx8;
typedef __attribute__((ext_vector_type(8)))  float  fx8;
typedef int intx4 __attribute__((vector_size(16)));

__device__ __forceinline__ unsigned short f2bf(float f) {
  union { float f; unsigned u; } v; v.f = f;
  unsigned u = v.u;
  return (unsigned short)((u + 0x7fffu + ((u >> 16) & 1u)) >> 16);
}

__device__ __forceinline__ float sigmoidf_(float x) {
  return 1.0f / (1.0f + __expf(-x));
}

// ---- async global -> LDS (ASYNCcnt path), builtin if present else raw asm ----
__device__ __forceinline__ void async_ld_b128(const unsigned short* g, unsigned char* l) {
#if __has_builtin(__builtin_amdgcn_global_load_async_to_lds_b128)
  __builtin_amdgcn_global_load_async_to_lds_b128((AS1 intx4*)g, (AS3 intx4*)l, 0, 0);
#else
  unsigned int lofs = (unsigned int)(unsigned long long)(AS3 unsigned char*)l;
  asm volatile("global_load_async_to_lds_b128 %0, %1, off"
               :: "v"(lofs), "v"(g) : "memory");
#endif
}

__device__ __forceinline__ void wait_async5() {
#if __has_builtin(__builtin_amdgcn_s_wait_asynccnt)
  __builtin_amdgcn_s_wait_asynccnt(5);
#else
  asm volatile("s_wait_asynccnt 0x5" ::: "memory");
#endif
}

__device__ __forceinline__ void wait_async0() {
#if __has_builtin(__builtin_amdgcn_s_wait_asynccnt)
  __builtin_amdgcn_s_wait_asynccnt(0);
#else
  asm volatile("s_wait_asynccnt 0x0" ::: "memory");
#endif
}

// ---------------- f32 -> bf16 convert ----------------
__global__ void __launch_bounds__(256)
k_cvt_bf16(const float* __restrict__ src, unsigned short* __restrict__ dst, int n) {
  int i = blockIdx.x * 256 + threadIdx.x;
  if (i < n) dst[i] = f2bf(src[i]);
}

// =====================================================================
// LDS-tiled WMMA GEMM: C[:, nOffset + 64*nBlock ...] = A[M,K] * W[N,K]^T
// Block = 256 threads (8 waves). Block tile = 256 (M) x 64 (N), k-step 32.
// Double-buffered async global->LDS (5 b128 chunks/thread/stage),
// s_wait_asynccnt + split barriers. Wave w computes rows [32w, 32w+32):
// two 16-row M-tiles sharing the 4 B fragments -> 8 WMMAs per k-step.
// =====================================================================
#define KS      32
#define APITCH  80               // 32 bf16 (64B) + 16B pad per row
#define MROWS   256
#define ABYTES  (MROWS * APITCH) // 20480
#define BBYTES  (64 * APITCH)    // 5120
#define STAGEB  (ABYTES + BBYTES)// 25600

__global__ void __launch_bounds__(256)
k_gemm_lds(const unsigned short* __restrict__ A,
           const unsigned short* __restrict__ W,
           float* __restrict__ C,
           int K, int nOffset, int ldc, int nBlocks) {
  __shared__ __align__(16) unsigned char sh[2 * STAGEB];
  const int tid    = threadIdx.x;
  const int wv     = tid >> 5;
  const int lane   = tid & 31;
  const int mBlock = blockIdx.x / nBlocks;
  const int nBlock = blockIdx.x - mBlock * nBlocks;
  const int nBase  = nOffset + (nBlock << 6);

  // per-thread async chunk assignment: 1280 16B chunks per stage, 5 per thread
  const unsigned short* gp[5];
  unsigned int lo[5];
  #pragma unroll
  for (int s2 = 0; s2 < 5; ++s2) {
    int c = tid + (s2 << 8);
    if (c < 1024) {                      // A tile: 256 rows x 4 chunks
      int row = c >> 2, ch = c & 3;
      gp[s2] = A + (size_t)(mBlock * MROWS + row) * K + ch * 8;
      lo[s2] = (unsigned)(row * APITCH + ch * 16);
    } else {                             // B tile: 64 rows x 4 chunks
      int c2 = c - 1024;
      int row = c2 >> 2, ch = c2 & 3;
      gp[s2] = W + (size_t)(nBase + row) * K + ch * 8;
      lo[s2] = (unsigned)(ABYTES + row * APITCH + ch * 16);
    }
  }

  // preload stage 0
  #pragma unroll
  for (int s2 = 0; s2 < 5; ++s2) async_ld_b128(gp[s2], sh + lo[s2]);

  const int half = lane >> 4;
  const int r    = lane & 15;
  fx8 a0c0 = {}; fx8 a0c1 = {}; fx8 a0c2 = {}; fx8 a0c3 = {};
  fx8 a1c0 = {}; fx8 a1c1 = {}; fx8 a1c2 = {}; fx8 a1c3 = {};

  const int steps = K / KS;
  for (int i = 0; i < steps; ++i) {
    const unsigned int cur = (unsigned)(i & 1) * STAGEB;
    const unsigned int nxt = STAGEB - cur;
    if (i + 1 < steps) {
      #pragma unroll
      for (int s2 = 0; s2 < 5; ++s2) {
        gp[s2] += KS;
        async_ld_b128(gp[s2], sh + nxt + lo[s2]);
      }
      wait_async5();       // previous stage's 5 copies complete (5 in flight)
    } else {
      wait_async0();
    }
    __syncthreads();       // stage published to all waves

    // A fragments (ISA 7.12.2): lanes<16 K{0..7,16..23}, lanes>=16 K{8..15,24..31}
    union { bfx16 v; bfx8 h[2]; } af0, af1;
    const unsigned char* aRow0 = sh + cur + (unsigned)((wv * 32 + r) * APITCH + half * 16);
    af0.h[0] = *(const bfx8*)(aRow0);
    af0.h[1] = *(const bfx8*)(aRow0 + 32);
    const unsigned char* aRow1 = aRow0 + 16 * APITCH;
    af1.h[0] = *(const bfx8*)(aRow1);
    af1.h[1] = *(const bfx8*)(aRow1 + 32);

    // B fragments (shared by both M-tiles): column n=r; same K order as A
    const unsigned char* bRow = sh + cur + (unsigned)(ABYTES + r * APITCH + half * 16);
    union { bfx16 v; bfx8 h[2]; } b0, b1, b2, b3;
    b0.h[0] = *(const bfx8*)(bRow);
    b0.h[1] = *(const bfx8*)(bRow + 32);
    b1.h[0] = *(const bfx8*)(bRow + 16 * APITCH);
    b1.h[1] = *(const bfx8*)(bRow + 16 * APITCH + 32);
    b2.h[0] = *(const bfx8*)(bRow + 32 * APITCH);
    b2.h[1] = *(const bfx8*)(bRow + 32 * APITCH + 32);
    b3.h[0] = *(const bfx8*)(bRow + 48 * APITCH);
    b3.h[1] = *(const bfx8*)(bRow + 48 * APITCH + 32);

    a0c0 = __builtin_amdgcn_wmma_f32_16x16x32_bf16(false, af0.v, false, b0.v,
                                                   (short)0, a0c0, false, false);
    a0c1 = __builtin_amdgcn_wmma_f32_16x16x32_bf16(false, af0.v, false, b1.v,
                                                   (short)0, a0c1, false, false);
    a0c2 = __builtin_amdgcn_wmma_f32_16x16x32_bf16(false, af0.v, false, b2.v,
                                                   (short)0, a0c2, false, false);
    a0c3 = __builtin_amdgcn_wmma_f32_16x16x32_bf16(false, af0.v, false, b3.v,
                                                   (short)0, a0c3, false, false);
    a1c0 = __builtin_amdgcn_wmma_f32_16x16x32_bf16(false, af1.v, false, b0.v,
                                                   (short)0, a1c0, false, false);
    a1c1 = __builtin_amdgcn_wmma_f32_16x16x32_bf16(false, af1.v, false, b1.v,
                                                   (short)0, a1c1, false, false);
    a1c2 = __builtin_amdgcn_wmma_f32_16x16x32_bf16(false, af1.v, false, b2.v,
                                                   (short)0, a1c2, false, false);
    a1c3 = __builtin_amdgcn_wmma_f32_16x16x32_bf16(false, af1.v, false, b3.v,
                                                   (short)0, a1c3, false, false);

    __syncthreads();       // everyone done reading before buffer is overwritten
  }

  // C store: VGPR rr -> M = rr + 8*half, N = lane&15
  float* cPtr0 = C + (size_t)(mBlock * MROWS + wv * 32 + half * 8) * ldc + nBase + r;
  float* cPtr1 = cPtr0 + (size_t)16 * ldc;
  #pragma unroll
  for (int rr = 0; rr < 8; ++rr) {
    cPtr0[(size_t)rr * ldc]      = a0c0[rr];
    cPtr0[(size_t)rr * ldc + 16] = a0c1[rr];
    cPtr0[(size_t)rr * ldc + 32] = a0c2[rr];
    cPtr0[(size_t)rr * ldc + 48] = a0c3[rr];
    cPtr1[(size_t)rr * ldc]      = a1c0[rr];
    cPtr1[(size_t)rr * ldc + 16] = a1c1[rr];
    cPtr1[(size_t)rr * ldc + 32] = a1c2[rr];
    cPtr1[(size_t)rr * ldc + 48] = a1c3[rr];
  }
}

// =====================================================================
// Simple branch-free WMMA GEMM (template NV tiles of 16 cols per wave).
// Used only for the 32-column tail of x_proj (N=96).
// =====================================================================
template <int NV>
__global__ void __launch_bounds__(256)
k_gemm_simple(const unsigned short* __restrict__ A,
              const unsigned short* __restrict__ W,
              float* __restrict__ C,
              int K, int nOffset, int ldc, int mTiles) {
  const int wave = (blockIdx.x * 256 + (int)threadIdx.x) >> 5;
  const int lane = threadIdx.x & 31;
  if (wave >= mTiles) return;
  const int half = lane >> 4;
  const int r    = lane & 15;

  const unsigned short* aPtr = A + (size_t)(wave * 16 + r) * K + half * 8;
  const unsigned short* wp[NV];
  #pragma unroll
  for (int j = 0; j < NV; ++j)
    wp[j] = W + (size_t)(nOffset + j * 16 + r) * K + half * 8;

  fx8 acc[NV];
  #pragma unroll
  for (int j = 0; j < NV; ++j) acc[j] = (fx8){};

  union { bfx16 v; bfx8 h[2]; } af, bf;
  for (int k0 = 0; k0 < K; k0 += 32) {
    af.h[0] = *reinterpret_cast<const bfx8*>(aPtr + k0);
    af.h[1] = *reinterpret_cast<const bfx8*>(aPtr + k0 + 16);
    #pragma unroll
    for (int j = 0; j < NV; ++j) {
      bf.h[0] = *reinterpret_cast<const bfx8*>(wp[j] + k0);
      bf.h[1] = *reinterpret_cast<const bfx8*>(wp[j] + k0 + 16);
      acc[j] = __builtin_amdgcn_wmma_f32_16x16x32_bf16(false, af.v, false, bf.v,
                                                       (short)0, acc[j], false, false);
    }
  }

  float* cPtr = C + (size_t)(wave * 16 + half * 8) * ldc + nOffset + r;
  #pragma unroll
  for (int rr = 0; rr < 8; ++rr) {
    #pragma unroll
    for (int j = 0; j < NV; ++j) cPtr[(size_t)rr * ldc + j * 16] = acc[j][rr];
  }
}

// ---------------- causal depthwise conv (K=4) + SiLU ----------------
__global__ void __launch_bounds__(256)
k_conv_silu(const float* __restrict__ proj, const float* __restrict__ cw,
            const float* __restrict__ cb, float* __restrict__ u,
            unsigned short* __restrict__ ub) {
  int i = blockIdx.x * 256 + threadIdx.x;
  if (i >= ML * DI) return;
  int d = i & (DI - 1);
  int m = i >> 11;
  int l = m & (LLEN - 1);
  const float* w = cw + d * 4;
  float s = cb[d];
  #pragma unroll
  for (int j = 0; j < 4; ++j) {
    int back = 3 - j;
    if (l >= back) s += w[j] * proj[(size_t)(m - back) * PROJW + d];
  }
  float val = s * sigmoidf_(s);
  u[i]  = val;
  ub[i] = f2bf(val);
}

// ---------------- extract delta_raw (cols 0..63 of p) as bf16 ----------------
__global__ void __launch_bounds__(256)
k_extract_dt(const float* __restrict__ p, unsigned short* __restrict__ draw) {
  int i = blockIdx.x * 256 + threadIdx.x;
  if (i >= ML * DTR) return;
  int m = i >> 6;
  int c = i & 63;
  draw[i] = f2bf(p[(size_t)m * PDIM + c]);
}

// ---------------- softplus(delta + bias), in place ----------------
__global__ void __launch_bounds__(256)
k_softplus(float* __restrict__ dl, const float* __restrict__ db) {
  int i = blockIdx.x * 256 + threadIdx.x;
  if (i >= ML * DI) return;
  float v = dl[i] + db[i & (DI - 1)];
  dl[i] = (v > 20.0f) ? v : __logf(1.0f + __expf(v));
}

// ---------------- selective scan: one lane per (b,d) channel ----------------
__global__ void __launch_bounds__(256)
k_scan(const float* __restrict__ delta, const float* __restrict__ u,
       const float* __restrict__ p, const float* __restrict__ A_log,
       float* __restrict__ y) {
  int t = blockIdx.x * 256 + threadIdx.x;
  if (t >= BB * DI) return;
  int b = t >> 11;
  int d = t & (DI - 1);
  float Aa[NS], st[NS];
  #pragma unroll
  for (int n = 0; n < NS; ++n) {
    Aa[n] = -__expf(A_log[d * NS + n]);
    st[n] = 0.0f;
  }
  for (int l = 0; l < LLEN; ++l) {
    size_t m = (size_t)b * LLEN + l;
    float dt = delta[m * DI + d];
    float uu = u[m * DI + d];
    const float* pr = p + m * PDIM;
    float bb[NS], cc[NS];
    #pragma unroll
    for (int q = 0; q < 4; ++q) {
      *reinterpret_cast<float4*>(bb + 4 * q) =
          reinterpret_cast<const float4*>(pr + DTR)[q];
      *reinterpret_cast<float4*>(cc + 4 * q) =
          reinterpret_cast<const float4*>(pr + DTR + NS)[q];
    }
    float du = dt * uu;
    float yv = 0.0f;
    #pragma unroll
    for (int n = 0; n < NS; ++n) {
      float dA = __expf(dt * Aa[n]);
      st[n] = dA * st[n] + du * bb[n];
      yv   += st[n] * cc[n];
    }
    y[m * DI + d] = yv;
  }
}

// ---------------- gate: (y + u*D) * silu(res) -> bf16 ----------------
__global__ void __launch_bounds__(256)
k_gate(const float* __restrict__ proj, const float* __restrict__ y,
       const float* __restrict__ u, const float* __restrict__ Dp,
       unsigned short* __restrict__ y2b) {
  int i = blockIdx.x * 256 + threadIdx.x;
  if (i >= ML * DI) return;
  int d = i & (DI - 1);
  int m = i >> 11;
  float res = proj[(size_t)m * PROJW + DI + d];
  float val = (y[i] + u[i] * Dp[d]) * (res * sigmoidf_(res));
  y2b[i] = f2bf(val);
}

extern "C" void kernel_launch(void* const* d_in, const int* in_sizes, int n_in,
                              void* d_out, int out_size, void* d_ws, size_t ws_size,
                              hipStream_t stream) {
  (void)in_sizes; (void)n_in; (void)out_size; (void)ws_size;
  const float* x     = (const float*)d_in[0];
  const float* w_in  = (const float*)d_in[1];
  const float* cw    = (const float*)d_in[2];
  const float* cb    = (const float*)d_in[3];
  const float* w_xp  = (const float*)d_in[4];
  const float* w_dt  = (const float*)d_in[5];
  const float* b_dt  = (const float*)d_in[6];
  const float* A_log = (const float*)d_in[7];
  const float* Dp    = (const float*)d_in[8];
  const float* w_out = (const float*)d_in[9];
  float* out = (float*)d_out;

  char*  ws  = (char*)d_ws;
  size_t off = 0;
  auto take = [&](size_t bytes) -> char* {
    char* pp = ws + off;
    off += (bytes + 255) & ~(size_t)255;
    return pp;
  };

  unsigned short* xb   = (unsigned short*)take((size_t)ML * DM * 2);
  unsigned short* w1b  = (unsigned short*)take((size_t)PROJW * DM * 2);
  unsigned short* xpb  = (unsigned short*)take((size_t)PDIM * DI * 2);
  unsigned short* dtwb = (unsigned short*)take((size_t)DI * DTR * 2);
  unsigned short* w3b  = (unsigned short*)take((size_t)DM * DI * 2);
  float*          proj = (float*)take((size_t)ML * PROJW * 4);
  float*          u    = (float*)take((size_t)ML * DI * 4);
  unsigned short* ub   = (unsigned short*)take((size_t)ML * DI * 2);
  float*          p    = (float*)take((size_t)ML * PDIM * 4);
  unsigned short* draw = (unsigned short*)take((size_t)ML * DTR * 2);
  float*          delt = (float*)take((size_t)ML * DI * 4);
  float*          ysc  = (float*)take((size_t)ML * DI * 4);
  unsigned short* y2b  = (unsigned short*)take((size_t)ML * DI * 2);

  auto cvt = [&](const float* s, unsigned short* dst, int n) {
    k_cvt_bf16<<<(n + 255) / 256, 256, 0, stream>>>(s, dst, n);
  };
  // LDS-tiled GEMM over a 64-column-multiple span starting at nOffset
  auto gemm = [&](const unsigned short* A, const unsigned short* W, float* Cc,
                  int K, int nOffset, int ldc, int nSpan) {
    int nBlocks = nSpan >> 6;
    int blocks  = (ML / MROWS) * nBlocks;
    k_gemm_lds<<<blocks, 256, 0, stream>>>(A, W, Cc, K, nOffset, ldc, nBlocks);
  };

  // precision staging
  cvt(x,     xb,   ML * DM);
  cvt(w_in,  w1b,  PROJW * DM);
  cvt(w_xp,  xpb,  PDIM * DI);
  cvt(w_dt,  dtwb, DI * DTR);
  cvt(w_out, w3b,  DM * DI);

  // in_proj: proj[4096,4096]
  gemm(xb, w1b, proj, DM, 0, PROJW, PROJW);
  // conv + silu -> u (f32 + bf16)
  k_conv_silu<<<(ML * DI + 255) / 256, 256, 0, stream>>>(proj, cw, cb, u, ub);
  // x_proj: p[4096,96] = 64-wide LDS launch + 32-wide simple tail
  gemm(ub, xpb, p, DI, 0, PDIM, 64);
  k_gemm_simple<2><<<(ML / 16 + 7) / 8, 256, 0, stream>>>(ub, xpb, p, DI, 64, PDIM, ML / 16);
  // delta_raw -> bf16
  k_extract_dt<<<(ML * DTR + 255) / 256, 256, 0, stream>>>(p, draw);
  // dt_proj: delt[4096,2048]
  gemm(draw, dtwb, delt, DTR, 0, DI, DI);
  // softplus(+bias) in place
  k_softplus<<<(ML * DI + 255) / 256, 256, 0, stream>>>(delt, b_dt);
  // selective scan (4096 channels)
  k_scan<<<(BB * DI + 255) / 256, 256, 0, stream>>>(delt, u, p, A_log, ysc);
  // gate -> bf16
  k_gate<<<(ML * DI + 255) / 256, 256, 0, stream>>>(proj, ysc, u, Dp, y2b);
  // out_proj -> d_out (f32)
  gemm(y2b, w3b, out, DI, 0, DM, DM);
}
